// BernsteinFlowModel_26499948216585
// MI455X (gfx1250) — matrix operand order repaired and checked by reading, not compiled, not run on hardware
//
#include <hip/hip_runtime.h>
#include <hip/hip_bf16.h>
#include <math.h>

typedef _Float16 v16h __attribute__((ext_vector_type(16)));
typedef _Float16 v8h  __attribute__((ext_vector_type(8)));
typedef _Float16 v4h  __attribute__((ext_vector_type(4)));
typedef float    v8f  __attribute__((ext_vector_type(8)));

// ---------------------------------------------------------------------------
// Workspace layout (f16 elements), c_alpha_i stored transposed+padded:
//   cT_i : [16 cols][K_i rows],  col 15 = zeros.   K_i = 4^i
//   offsets: cT0@0(16) cT1@16(64) cT2@80(256) cT3@336(1024) cT4@1360(4096)
//            cT5@5456(16384)  -> total 21840 f16 = 43680 bytes
// ---------------------------------------------------------------------------

__global__ __launch_bounds__(256)
void bflow_prep(const float* __restrict__ A0, const float* __restrict__ A1,
                const float* __restrict__ A2, const float* __restrict__ A3,
                const float* __restrict__ A4, const float* __restrict__ A5,
                _Float16* __restrict__ cw)
{
    int t = blockIdx.x * blockDim.x + threadIdx.x;
    if (t >= 1365) return;                       // 1+4+16+64+256+1024 rows total
    const int   Ks[6]   = {1, 4, 16, 64, 256, 1024};
    const int   rs[6]   = {0, 1, 5, 21, 85, 341};
    const int   offs[6] = {0, 16, 80, 336, 1360, 5456};
    const float* A = A0; int i = 0;
    if (t >= rs[5])      { i = 5; A = A5; }
    else if (t >= rs[4]) { i = 4; A = A4; }
    else if (t >= rs[3]) { i = 3; A = A3; }
    else if (t >= rs[2]) { i = 2; A = A2; }
    else if (t >= rs[1]) { i = 1; A = A1; }
    int r = t - rs[i];
    float c = 0.0f;
    for (int col = 0; col < 15; col++) {
        float a  = A[r * 15 + col];
        float sp = (a > 20.0f) ? a : log1pf(expf(a));   // softplus
        c += sp;
        // 2*(sigmoid(c)-0.5) == tanh(c/2)
        cw[offs[i] + col * Ks[i] + r] = (_Float16)tanhf(0.5f * c);
    }
    cw[offs[i] + 15 * Ks[i] + r] = (_Float16)0.0f;      // pad column 15
}

// ---------------------------------------------------------------------------
// Main kernel: one wave32 per 16-point tile.
// ---------------------------------------------------------------------------

__device__ __forceinline__ void calc_psi(float t, float* psi, float& phi15)
{
    // phi = degree-15 Bernstein basis scaled by 16; psi[j] = phi[j]-phi[j+1]
    const float comb15[16] = {1.f,15.f,105.f,455.f,1365.f,3003.f,5005.f,6435.f,
                              6435.f,5005.f,3003.f,1365.f,455.f,105.f,15.f,1.f};
    float u = 1.0f - t;
    float xp[16], up[16];
    xp[0] = 1.0f; up[0] = 1.0f;
    for (int a = 1; a < 16; a++) { xp[a] = xp[a-1] * t; up[a] = up[a-1] * u; }
    float phi[16];
    for (int a = 0; a < 16; a++) phi[a] = 16.0f * comb15[a] * xp[a] * up[15 - a];
    for (int j = 0; j < 15; j++) psi[j] = phi[j] - phi[j + 1];
    phi15 = phi[15];
}

__global__ __launch_bounds__(32)
void bflow_main(const float* __restrict__ x, const _Float16* __restrict__ cw,
                float* __restrict__ out)
{
    // ping-pong cond buffers (f16), per-point row stride = buffer capacity
    __shared__ alignas(16) _Float16 condA[16 * 256];    //  8 KB (holds up to K=256)
    __shared__ alignas(16) _Float16 condB[16 * 1024];   // 32 KB (holds up to K=1024)
    __shared__ float tf_lds[16][17];                    // GEMM result tile (+pad)
    __shared__ float dens[16];

    const int lane = threadIdx.x;
    const int p    = lane & 15;        // point within tile / matrix row / B column
    const int half = lane >> 4;        // wave half selects K sub-ranges
    const int g0   = blockIdx.x * 16;

    // ---- per-point inputs & degree-3 bases for x0..x4 ----------------------
    float xv[6];
    for (int j = 0; j < 6; j++) xv[j] = x[(g0 + p) * 6 + j];
    float b3[5][4];
    for (int j = 0; j < 5; j++) {
        float t = xv[j], u = 1.0f - t;
        b3[j][0] = u * u * u;
        b3[j][1] = 3.0f * t * u * u;
        b3[j][2] = 3.0f * t * t * u;
        b3[j][3] = t * t * t;
    }

    const bool owner = (half == 0);    // lanes 0..15 each own one point
    float psi[15], phi15;

    if (owner) {
        float d = 1.0f;
        // ---- dim 0 : K = 1 (scalar) ----
        calc_psi(xv[0], psi, phi15);
        float f0 = phi15;
        for (int c = 0; c < 15; c++) f0 += (float)cw[c] * psi[c];   // cT0[c]
        d *= f0;
        // ---- dim 1 : K = 4 (scalar) ----
        calc_psi(xv[1], psi, phi15);
        const _Float16* c1 = cw + 16;                 // cT1: [16][4]
        float f1 = phi15;
        for (int c = 0; c < 15; c++) {
            float tf = 0.0f;
            for (int k = 0; k < 4; k++) tf += b3[0][k] * (float)c1[c * 4 + k];
            f1 += tf * psi[c];
        }
        d *= f1;
        dens[p] = d;
        // ---- seed cond2 (K = 16) into condA, zero-pad K 16..31 for WMMA ----
        for (int m = 0; m < 16; m++)
            condA[p * 256 + m] = (_Float16)(b3[0][m >> 2] * b3[1][m & 3]);
        for (int m = 16; m < 32; m++)
            condA[p * 256 + m] = (_Float16)0.0f;
    }
    __syncthreads();

    // ---- dims 2..5 : WMMA GEMMs over K = {16,64,256,1024} ------------------
    const int cwOff[4] = {80, 336, 1360, 5456};
    const int Ks[4]    = {16, 64, 256, 1024};
    int cur = 0;                       // 0 -> condA is source, 1 -> condB

    const int koffA  = half * 16;      // A frag: K 0..15 vs 16..31
    const int kbaseB = half * 16;      // B frag: K 0..15 vs 16..31

    for (int ii = 0; ii < 4; ii++) {
        const int dim    = ii + 2;
        const int K      = Ks[ii];
        const int chunks = (K + 31) >> 5;
        const _Float16* cT = cw + cwOff[ii];
        _Float16* src = cur ? condB : condA;          // runtime select (no
        const int st  = cur ? 1024 : 256;             //  static initializer)

        v8f acc = {0.f, 0.f, 0.f, 0.f, 0.f, 0.f, 0.f, 0.f};
        for (int kc = 0; kc < chunks; kc++) {
            // A fragment (16x32 f16): per-lane two contiguous b128 LDS loads
            const _Float16* ap = &src[p * st + kc * 32 + koffA];
            v8h a0 = *(const v8h*)ap;
            v8h a1 = *(const v8h*)(ap + 8);
            v16h Af = __builtin_shufflevector(a0, a1,
                        0,1,2,3,4,5,6,7,8,9,10,11,12,13,14,15);
            // B fragment (32x16 f16): lane = column p, 16 consecutive K values
            const _Float16* bp = &cT[p * K + kc * 32 + kbaseB];
            v8h b0 = *(const v8h*)bp;
            v8h b1 = *(const v8h*)(bp + 8);
            v16h Bf = __builtin_shufflevector(b0, b1,
                        0,1,2,3,4,5,6,7,8,9,10,11,12,13,14,15);
            acc = __builtin_amdgcn_wmma_f32_16x16x32_f16(
                      false, Af, false, Bf, (short)0, acc, false, false);
        }
        // scatter D tile: lane holds column p of rows (v + 8*half)
        for (int v = 0; v < 8; v++) tf_lds[v + 8 * half][p] = acc[v];
        __syncthreads();

        if (owner) {
            calc_psi(xv[dim], psi, phi15);
            float f = phi15;
            for (int c = 0; c < 15; c++) f += tf_lds[p][c] * psi[c];
            dens[p] *= f;
        }
        __syncthreads();

        // Kronecker expansion cond_{dim} (x) B3(x_dim) -> other buffer.
        // Both wave halves split m by parity (distinct dst ranges, no races).
        if (ii < 3) {
            _Float16* dst = cur ? condA : condB;
            const int dstStride = cur ? 256 : 1024;
            for (int m = half; m < K; m += 2) {
                float cv = (float)src[p * st + m];
                v4h w = { (_Float16)(cv * b3[dim][0]),
                          (_Float16)(cv * b3[dim][1]),
                          (_Float16)(cv * b3[dim][2]),
                          (_Float16)(cv * b3[dim][3]) };
                *(v4h*)&dst[p * dstStride + 4 * m] = w;
            }
            cur ^= 1;
        }
        __syncthreads();
    }

    if (owner) out[g0 + p] = dens[p];
}

// ---------------------------------------------------------------------------
extern "C" void kernel_launch(void* const* d_in, const int* in_sizes, int n_in,
                              void* d_out, int out_size, void* d_ws, size_t ws_size,
                              hipStream_t stream)
{
    const float* x  = (const float*)d_in[0];
    const float* A0 = (const float*)d_in[1];
    const float* A1 = (const float*)d_in[2];
    const float* A2 = (const float*)d_in[3];
    const float* A3 = (const float*)d_in[4];
    const float* A4 = (const float*)d_in[5];
    const float* A5 = (const float*)d_in[6];
    float*     out = (float*)d_out;
    _Float16*  cw  = (_Float16*)d_ws;      // 21840 f16 = 43680 bytes used

    bflow_prep<<<6, 256, 0, stream>>>(A0, A1, A2, A3, A4, A5, cw);

    const int N = in_sizes[0] / 6;         // 65536 points
    bflow_main<<<N / 16, 32, 0, stream>>>(x, cw, out);
}
